// MiniGPT_78022375899521
// MI455X (gfx1250) — matrix operand order repaired and checked by reading, not compiled, not run on hardware
//
#include <hip/hip_runtime.h>
#include <hip/hip_bf16.h>

#define Tt   1024
#define Bb   4
#define Dd   768
#define Hh   12
#define HDc  64
#define Ll   4
#define Vv   32000
#define S3c  2304            // 3*D, row stride of fused qkv buffer
#define Mrows (Bb*Tt)        // 4096
#define D4c  (4*Dd)          // 3072

typedef unsigned short u16;
typedef __attribute__((ext_vector_type(16))) __bf16 v16bf;
typedef __attribute__((ext_vector_type(8)))  float  v8f;
typedef __attribute__((ext_vector_type(4)))  unsigned int u32x4;
typedef __attribute__((ext_vector_type(8)))  unsigned int u32x8;

union FragBF { uint4 u[2]; v16bf v; };

static __device__ __forceinline__ u16 f2bf(float f) {
  unsigned u = __float_as_uint(f);
  unsigned r = u + 0x7FFFu + ((u >> 16) & 1u);   // round-to-nearest-even
  return (u16)(r >> 16);
}
static __device__ __forceinline__ v8f vzero() {
  v8f z;
#pragma unroll
  for (int i = 0; i < 8; ++i) z[i] = 0.f;
  return z;
}

#define WMMA_BF16(A, B, C) \
  __builtin_amdgcn_wmma_f32_16x16x32_bf16(false, (A), false, (B), (short)0, (C), false, false)

// ---------------------------------------------------------------------------
// TDM: DMA one 128-row x 32-bf16 tile (row stride K elems) from global into
// LDS at byte offset ldsOff, with LDS padding 16 B after every 64 B stored
// => padded LDS row stride of 80 B = 40 bf16 (matches WMMA fragment reads).
// Descriptor per CDNA5 ISA ch.8 (D# group0 128b, group1 256b); VADDR2/3=NULL.
// Wave-scalar op, tracked by TENSORcnt.
// ---------------------------------------------------------------------------
static __device__ __forceinline__ void tdm_load_tile(const u16* gbase,
                                                     unsigned ldsOff, int K) {
  unsigned long long ga = (unsigned long long)gbase;
  u32x4 g0;
  g0[0] = 1u;                                           // count=1, user mode
  g0[1] = ldsOff;                                       // lds_addr (bytes)
  g0[2] = (unsigned)ga;                                 // global_addr[31:0]
  g0[3] = (unsigned)((ga >> 32) & 0x1FFFFFFu) | (2u << 30); // [56:32] | type=2
  u32x8 g1;
  g1[0] = (1u << 16) | (1u << 20) | (3u << 22) | (3u << 25);
         // data_size=2B | pad_enable | pad_interval=16dw | pad_amount=4dw
  g1[1] = ((unsigned)K & 0xFFFFu) << 16;                // tensor_dim0[15:0]
  g1[2] = (128u << 16) | ((unsigned)K >> 16);           // dim1[15:0] | dim0[31:16]
  g1[3] = (32u << 16);                                  // tile_dim0=32
  g1[4] = 128u;                                         // tile_dim1=128, tile_dim2=0
  g1[5] = (unsigned)K;                                  // tensor_dim0_stride[31:0]
  g1[6] = 0u;                                           // stride hi / dim1_stride lo
  g1[7] = 0u;
  asm volatile("tensor_load_to_lds %0, %1" :: "s"(g0), "s"(g1) : "memory");
}

// ---------------------------------------------------------------------------
// Embedding gather: x[b*T+t, :] = tok_emb[idx[b,t], :] + pos_emb[t, :]
// ---------------------------------------------------------------------------
__global__ __launch_bounds__(256) void embed_kernel(
    const int* __restrict__ idx, const float* __restrict__ tok,
    const float* __restrict__ pos, float* __restrict__ x) {
  int row = blockIdx.x;                  // b*T + t
  int t   = row & (Tt - 1);
  int tk  = idx[row];
  const float* tp = tok + (size_t)tk * Dd;
  const float* pp = pos + (size_t)t * Dd;
  float* xp = x + (size_t)row * Dd;
  for (int c = threadIdx.x; c < Dd; c += 256) xp[c] = tp[c] + pp[c];
}

// ---------------------------------------------------------------------------
// LayerNorm over D=768, one block per row, bf16 output for WMMA A-operand
// ---------------------------------------------------------------------------
__global__ __launch_bounds__(256) void ln_kernel(
    const float* __restrict__ x, const float* __restrict__ g,
    const float* __restrict__ b, u16* __restrict__ out) {
  int row = blockIdx.x;
  const float* xr = x + (size_t)row * Dd;
  float s = 0.f, s2 = 0.f;
  for (int c = threadIdx.x; c < Dd; c += 256) { float v = xr[c]; s += v; s2 += v * v; }
#pragma unroll
  for (int m = 16; m > 0; m >>= 1) { s += __shfl_xor(s, m, 32); s2 += __shfl_xor(s2, m, 32); }
  __shared__ float sm[8], sv[8];
  if ((threadIdx.x & 31) == 0) { sm[threadIdx.x >> 5] = s; sv[threadIdx.x >> 5] = s2; }
  __syncthreads();
  __shared__ float mean_s, rstd_s;
  if (threadIdx.x < 8) {
    float ts = sm[threadIdx.x], ts2 = sv[threadIdx.x];
#pragma unroll
    for (int m = 4; m > 0; m >>= 1) { ts += __shfl_xor(ts, m, 32); ts2 += __shfl_xor(ts2, m, 32); }
    if (threadIdx.x == 0) {
      float mean = ts / (float)Dd;
      float var  = ts2 / (float)Dd - mean * mean;
      mean_s = mean; rstd_s = rsqrtf(var + 1e-5f);
    }
  }
  __syncthreads();
  float mean = mean_s, r = rstd_s;
  u16* op = out + (size_t)row * Dd;
  for (int c = threadIdx.x; c < Dd; c += 256)
    op[c] = f2bf((xr[c] - mean) * r * g[c] + b[c]);
}

// ---------------------------------------------------------------------------
// Transpose + fp32->bf16 convert: in[K,N] fp32 -> out[N,K] bf16. 32x32 tiles.
// ---------------------------------------------------------------------------
__global__ __launch_bounds__(256) void tconv_kernel(
    const float* __restrict__ in, u16* __restrict__ out, int K, int N) {
  __shared__ float tile[32][33];
  int tx = threadIdx.x & 31, ty = threadIdx.x >> 5;  // 32 x 8
  int n0 = blockIdx.x * 32, k0 = blockIdx.y * 32;
#pragma unroll
  for (int r = 0; r < 4; ++r)
    tile[ty + r * 8][tx] = in[(size_t)(k0 + ty + r * 8) * N + n0 + tx];
  __syncthreads();
#pragma unroll
  for (int r = 0; r < 4; ++r)
    out[(size_t)(n0 + ty + r * 8) * K + k0 + tx] = f2bf(tile[tx][ty + r * 8]);
}

// ---------------------------------------------------------------------------
// Vt[b,h,hd,t] = V[b*T+t, 2D + h*64+hd]  (bf16 -> bf16, coalesced writes)
// ---------------------------------------------------------------------------
__global__ __launch_bounds__(256) void vtrans_kernel(
    const u16* __restrict__ qkv, u16* __restrict__ vt) {
  int i = blockIdx.x * 256 + threadIdx.x;       // over B*H*HD*T
  int t  = i & (Tt - 1);
  int r  = i >> 10;
  int hd = r & (HDc - 1);
  int bh = r >> 6;
  int b = bh / Hh, h = bh % Hh;
  vt[i] = qkv[(size_t)(b * Tt + t) * S3c + 2 * Dd + h * HDc + hd];
}

// ---------------------------------------------------------------------------
// Generic bf16 WMMA GEMM:  C[M,N] = A[M,K] * Bt[N,K]^T  (+bias)(+relu)(+resid)
// Block 128x128, 8 waves of 32x64. Tiles staged by the Tensor Data Mover,
// double-buffered on TENSORcnt; K-loop manually unrolled by 64 so both
// pipeline stages use static LDS addresses. K % 64 == 0 for all call sites.
// ---------------------------------------------------------------------------
__global__ __launch_bounds__(256, 1) void gemm_bf16_kernel(
    const u16* __restrict__ A, const u16* __restrict__ Bt,
    const float* __restrict__ bias, const float* __restrict__ resid,
    float* __restrict__ outF, u16* __restrict__ outB,
    int M, int N, int K, int doRelu) {
  __shared__ __align__(16) u16 sA[2][128 * 40];   // 10240 B per buffer
  __shared__ __align__(16) u16 sB[2][128 * 40];
  const int tid  = threadIdx.x;
  const int lane = tid & 31, wave = tid >> 5;
  const int wm = wave >> 1, wn = wave & 1;               // 4 x 2 wave grid
  const int m0 = blockIdx.y * 128, n0 = blockIdx.x * 128;
  const int lm = lane & 15, hi = lane >> 4;
  const int kbA = hi ? 8 : 0;                            // ISA 16-bit A frag split
  const int kbB = hi ? 8 : 0;                            // ISA 16-bit B frag split

  const unsigned ldsA0 = (unsigned)(unsigned long long)&sA[0][0];
  const unsigned ldsB0 = (unsigned)(unsigned long long)&sB[0][0];
  const u16* gA = A + (size_t)m0 * K;
  const u16* gB = Bt + (size_t)n0 * K;

  v8f acc[2][4];
#pragma unroll
  for (int i = 0; i < 2; ++i)
#pragma unroll
    for (int j = 0; j < 4; ++j) acc[i][j] = vzero();

  // one WMMA K-step from a statically-addressed buffer pair
  auto compute_tile = [&](const u16* bA, const u16* bB) {
    FragBF fa[2], fb[4];
#pragma unroll
    for (int mt = 0; mt < 2; ++mt) {
      int r = wm * 32 + mt * 16 + lm;
      fa[mt].u[0] = *(const uint4*)(bA + r * 40 + kbA);
      fa[mt].u[1] = *(const uint4*)(bA + r * 40 + kbA + 16);
    }
#pragma unroll
    for (int nt = 0; nt < 4; ++nt) {
      int n = wn * 64 + nt * 16 + lm;
      fb[nt].u[0] = *(const uint4*)(bB + n * 40 + kbB);
      fb[nt].u[1] = *(const uint4*)(bB + n * 40 + kbB + 16);
    }
#pragma unroll
    for (int mt = 0; mt < 2; ++mt)
#pragma unroll
      for (int nt = 0; nt < 4; ++nt)
        acc[mt][nt] = WMMA_BF16(fa[mt].v, fb[nt].v, acc[mt][nt]);
  };

  if (wave == 0) {                        // prologue: fill buffer 0 (k=0)
    tdm_load_tile(gA, ldsA0, K);
    tdm_load_tile(gB, ldsB0, K);
  }

  for (int k0 = 0; k0 < K; k0 += 64) {    // K % 64 == 0
    // ---- stage 0: compute buf0 (k0), TDM prefetch buf1 (k0+32) ----
    if (wave == 0) {
      tdm_load_tile(gA + k0 + 32, ldsA0 + 10240u, K);   // k0+32 < K always
      tdm_load_tile(gB + k0 + 32, ldsB0 + 10240u, K);
      __builtin_amdgcn_s_wait_tensorcnt((short)2);      // buf0 complete
    }
    __syncthreads();
    compute_tile(&sA[0][0], &sB[0][0]);
    __syncthreads();

    // ---- stage 1: compute buf1 (k0+32), TDM prefetch buf0 (k0+64) ----
    if (wave == 0) {
      if (k0 + 64 < K) {
        tdm_load_tile(gA + k0 + 64, ldsA0, K);
        tdm_load_tile(gB + k0 + 64, ldsB0, K);
        __builtin_amdgcn_s_wait_tensorcnt((short)2);    // buf1 complete
      } else {
        __builtin_amdgcn_s_wait_tensorcnt((short)0);
      }
    }
    __syncthreads();
    compute_tile(&sA[1][0], &sB[1][0]);
    __syncthreads();
  }

  // Epilogue: bias / relu / residual / fp32 and-or bf16 stores
#pragma unroll
  for (int mt = 0; mt < 2; ++mt)
#pragma unroll
    for (int nt = 0; nt < 4; ++nt) {
      int n = n0 + wn * 64 + nt * 16 + lm;
      float bv = bias ? bias[n] : 0.f;
#pragma unroll
      for (int j = 0; j < 8; ++j) {
        int m = m0 + wm * 32 + mt * 16 + j + 8 * hi;
        float v = acc[mt][nt][j] + bv;
        if (doRelu) v = fmaxf(v, 0.f);
        size_t o = (size_t)m * N + n;
        if (resid) v += resid[o];
        if (outF) outF[o] = v;
        if (outB) outB[o] = f2bf(v);
      }
    }
}

// ---------------------------------------------------------------------------
// Causal flash attention. Grid (T/128, B*H), 8 waves; wave owns 16 q rows.
// S = Q K^T via 2 WMMAs; online softmax; P bounced through wave-private LDS
// (C-layout -> A-layout); O += P V via 4 WMMAs against pre-transposed Vt.
// ---------------------------------------------------------------------------
__global__ __launch_bounds__(256, 1) void attn_kernel(
    const u16* __restrict__ qkv, const u16* __restrict__ vt,
    u16* __restrict__ att) {
  __shared__ __align__(16) u16 sP[8 * 16 * 40];
  const int tid = threadIdx.x, wave = tid >> 5, lane = tid & 31;
  const int bh = blockIdx.y, b = bh / Hh, h = bh % Hh;
  const int qt = blockIdx.x * 128 + wave * 16;
  const int lm = lane & 15, hi = lane >> 4;
  const int kbA = hi ? 8 : 0;
  const int kb2 = hi ? 8 : 0;
  u16* wp = sP + wave * 640;                      // 16 x 40 wave-private tile

  // Q A-fragments (row qt+lm, hd 0..31 and 32..63)
  const u16* qp = qkv + (size_t)(b * Tt + qt + lm) * S3c + h * HDc;
  FragBF aq0, aq1;
  aq0.u[0] = *(const uint4*)(qp + kbA);
  aq0.u[1] = *(const uint4*)(qp + kbA + 16);
  aq1.u[0] = *(const uint4*)(qp + 32 + kbA);
  aq1.u[1] = *(const uint4*)(qp + 32 + kbA + 16);

  v8f oacc[4];
#pragma unroll
  for (int vI = 0; vI < 4; ++vI) oacc[vI] = vzero();
  float rmax[8], rsum[8];
#pragma unroll
  for (int j = 0; j < 8; ++j) { rmax[j] = -1e30f; rsum[j] = 0.f; }

  const int cmax = (qt + 15) >> 5;                // last 32-key chunk (causal)
  for (int c = 0; c <= cmax; ++c) {
    int k0 = c * 32;
    v8f s[2];
#pragma unroll
    for (int st = 0; st < 2; ++st) {
      int key = k0 + st * 16 + lm;                // B-frag column = key index
      const u16* kp = qkv + (size_t)(b * Tt + key) * S3c + Dd + h * HDc;
      FragBF bk0, bk1;
      bk0.u[0] = *(const uint4*)(kp + kb2);
      bk0.u[1] = *(const uint4*)(kp + kb2 + 16);
      bk1.u[0] = *(const uint4*)(kp + 32 + kb2);
      bk1.u[1] = *(const uint4*)(kp + 32 + kb2 + 16);
      v8f a = vzero();
      a = WMMA_BF16(aq0.v, bk0.v, a);
      a = WMMA_BF16(aq1.v, bk1.v, a);
      s[st] = a;
    }
    // scale, causal mask, online softmax update (row = qt + j + 8*hi)
#pragma unroll
    for (int j = 0; j < 8; ++j) {
      int m = j + 8 * hi;
      int qrow = qt + m;
      float a0 = s[0][j] * 0.125f; if (k0 + lm      > qrow) a0 = -1e30f;
      float a1 = s[1][j] * 0.125f; if (k0 + 16 + lm > qrow) a1 = -1e30f;
      float mx = fmaxf(a0, a1);
#pragma unroll
      for (int d = 1; d < 16; d <<= 1) mx = fmaxf(mx, __shfl_xor(mx, d, 32));
      float nm = fmaxf(rmax[j], mx);
      float corr = __expf(rmax[j] - nm);
      rmax[j] = nm;
      float p0 = __expf(a0 - nm), p1 = __expf(a1 - nm);
      float ps = p0 + p1;
#pragma unroll
      for (int d = 1; d < 16; d <<= 1) ps += __shfl_xor(ps, d, 32);
      rsum[j] = rsum[j] * corr + ps;
      wp[m * 40 + lm]      = f2bf(p0);
      wp[m * 40 + 16 + lm] = f2bf(p1);
#pragma unroll
      for (int vI = 0; vI < 4; ++vI) oacc[vI][j] *= corr;
    }
    // same-wave DS ops are in-order; fence to be explicit + block reordering
    asm volatile("s_wait_dscnt 0" ::: "memory");
    // P as 16x32 A-fragment
    FragBF ap;
    ap.u[0] = *(const uint4*)(wp + lm * 40 + kbA);
    ap.u[1] = *(const uint4*)(wp + lm * 40 + kbA + 16);
    // O += P * V : Vt[bh, hd, t] rows are key-contiguous => B-frag ready
#pragma unroll
    for (int vI = 0; vI < 4; ++vI) {
      const u16* vp = vt + ((size_t)bh * HDc + vI * 16 + lm) * Tt + k0 + kb2;
      FragBF bv;
      bv.u[0] = *(const uint4*)(vp);
      bv.u[1] = *(const uint4*)(vp + 16);
      oacc[vI] = WMMA_BF16(ap.v, bv.v, oacc[vI]);
    }
  }
  // normalize and store attn output (bf16, [B*T, D])
#pragma unroll
  for (int j = 0; j < 8; ++j) {
    int m = j + 8 * hi;
    float inv = 1.f / rsum[j];
    size_t rowo = (size_t)(b * Tt + qt + m) * Dd + h * HDc;
#pragma unroll
    for (int vI = 0; vI < 4; ++vI)
      att[rowo + vI * 16 + lm] = f2bf(oacc[vI][j] * inv);
  }
}

// ---------------------------------------------------------------------------
extern "C" void kernel_launch(void* const* d_in, const int* in_sizes, int n_in,
                              void* d_out, int out_size, void* d_ws, size_t ws_size,
                              hipStream_t stream) {
  const int*   idx    = (const int*)d_in[0];
  const float* tok    = (const float*)d_in[1];
  const float* pos    = (const float*)d_in[2];
  const float* wq     = (const float*)d_in[3];
  const float* wk     = (const float*)d_in[4];
  const float* wv     = (const float*)d_in[5];
  const float* wo     = (const float*)d_in[6];
  const float* bo     = (const float*)d_in[7];
  const float* ln1_g  = (const float*)d_in[8];
  const float* ln1_b  = (const float*)d_in[9];
  const float* ln2_g  = (const float*)d_in[10];
  const float* ln2_b  = (const float*)d_in[11];
  const float* w1     = (const float*)d_in[12];
  const float* b1     = (const float*)d_in[13];
  const float* w2     = (const float*)d_in[14];
  const float* b2     = (const float*)d_in[15];
  const float* lnf_g  = (const float*)d_in[16];
  const float* lnf_b  = (const float*)d_in[17];
  const float* w_lm   = (const float*)d_in[18];
  const float* b_lm   = (const float*)d_in[19];
  float* out = (float*)d_out;

  char* base = (char*)d_ws;
  size_t off = 0;
  auto take = [&](size_t bytes) -> void* {
    void* p = base + off;
    off += (bytes + 255) & ~(size_t)255;
    return p;
  };
  float* xf     = (float*)take((size_t)Mrows * Dd * 4);
  u16*   h_bf   = (u16*)take((size_t)Mrows * Dd * 2);
  u16*   qkv_bf = (u16*)take((size_t)Mrows * S3c * 2);
  u16*   vt_bf  = (u16*)take((size_t)Bb * Hh * HDc * Tt * 2);
  u16*   att_bf = (u16*)take((size_t)Mrows * Dd * 2);
  u16*   mlp_bf = (u16*)take((size_t)Mrows * D4c * 2);
  u16*   wqkvT  = (u16*)take((size_t)S3c * Dd * 2);
  u16*   woT    = (u16*)take((size_t)Dd * Dd * 2);
  u16*   w1T    = (u16*)take((size_t)D4c * Dd * 2);
  u16*   w2T    = (u16*)take((size_t)Dd * D4c * 2);
  u16*   wlmT   = (u16*)take((size_t)Vv * Dd * 2);

  embed_kernel<<<Mrows, 256, 0, stream>>>(idx, tok, pos, xf);

  for (int l = 0; l < Ll; ++l) {
    ln_kernel<<<Mrows, 256, 0, stream>>>(xf, ln1_g + l * Dd, ln1_b + l * Dd, h_bf);
    // fused QKV weight: rows [0,D)=Wq^T, [D,2D)=Wk^T, [2D,3D)=Wv^T, each [D,K=D]
    tconv_kernel<<<dim3(Dd / 32, Dd / 32), 256, 0, stream>>>(wq + (size_t)l * Dd * Dd, wqkvT, Dd, Dd);
    tconv_kernel<<<dim3(Dd / 32, Dd / 32), 256, 0, stream>>>(wk + (size_t)l * Dd * Dd, wqkvT + (size_t)Dd * Dd, Dd, Dd);
    tconv_kernel<<<dim3(Dd / 32, Dd / 32), 256, 0, stream>>>(wv + (size_t)l * Dd * Dd, wqkvT + (size_t)2 * Dd * Dd, Dd, Dd);
    gemm_bf16_kernel<<<dim3(S3c / 128, Mrows / 128), 256, 0, stream>>>(
        h_bf, wqkvT, nullptr, nullptr, nullptr, qkv_bf, Mrows, S3c, Dd, 0);
    vtrans_kernel<<<(Bb * Hh * HDc * Tt) / 256, 256, 0, stream>>>(qkv_bf, vt_bf);
    attn_kernel<<<dim3(Tt / 128, Bb * Hh), 256, 0, stream>>>(qkv_bf, vt_bf, att_bf);
    tconv_kernel<<<dim3(Dd / 32, Dd / 32), 256, 0, stream>>>(wo + (size_t)l * Dd * Dd, woT, Dd, Dd);
    gemm_bf16_kernel<<<dim3(Dd / 128, Mrows / 128), 256, 0, stream>>>(
        att_bf, woT, bo + l * Dd, xf, xf, nullptr, Mrows, Dd, Dd, 0);   // x += o@wo + bo
    ln_kernel<<<Mrows, 256, 0, stream>>>(xf, ln2_g + l * Dd, ln2_b + l * Dd, h_bf);
    tconv_kernel<<<dim3(D4c / 32, Dd / 32), 256, 0, stream>>>(w1 + (size_t)l * Dd * D4c, w1T, Dd, D4c);
    gemm_bf16_kernel<<<dim3(D4c / 128, Mrows / 128), 256, 0, stream>>>(
        h_bf, w1T, b1 + l * D4c, nullptr, nullptr, mlp_bf, Mrows, D4c, Dd, 1);  // relu
    tconv_kernel<<<dim3(Dd / 32, D4c / 32), 256, 0, stream>>>(w2 + (size_t)l * D4c * Dd, w2T, D4c, Dd);
    gemm_bf16_kernel<<<dim3(Dd / 128, Mrows / 128), 256, 0, stream>>>(
        mlp_bf, w2T, b2 + l * Dd, xf, xf, nullptr, Mrows, Dd, D4c, 0);  // x += mlp + b2
  }

  ln_kernel<<<Mrows, 256, 0, stream>>>(xf, lnf_g, lnf_b, h_bf);
  tconv_kernel<<<dim3(Vv / 32, Dd / 32), 256, 0, stream>>>(w_lm, wlmT, Dd, Vv);
  gemm_bf16_kernel<<<dim3(Vv / 128, Mrows / 128), 256, 0, stream>>>(
      h_bf, wlmT, b_lm, nullptr, out, nullptr, Mrows, Vv, Dd, 0);       // logits fp32
}
